// MultiHeadAttention_43379169690396
// MI455X (gfx1250) — matrix operand order repaired and checked by reading, not compiled, not run on hardware
//
#include <hip/hip_runtime.h>
#include <hip/hip_bf16.h>

// ---------------------------------------------------------------------------
// MHA forward for MI455X (gfx1250, wave32, WMMA bf16 16x16x32, f32 accum).
//   B=2, S=2048, D=1024, H=16, Dh=64, causal.
// Pipeline:
//   1) cvt: Wq/Wk/Wv/Wo fp32 -> bf16 (workspace)
//   2) gemm_wmma<MODE_QK>: Q = q@Wq^T+b -> Qh[bh][s][64]  (bf16)
//                          K = k@Wk^T+b -> Kh[bh][s][64]  (bf16)
//      gemm_wmma<MODE_VT>: V = v@Wv^T+b -> Vt[bh][64][s]  (bf16, transposed)
//   3) attn_wmma: flash-style causal attention, 1 wave per (bh, 16-q tile)
//                 -> AO[token][1024] bf16
//   4) gemm_wmma<MODE_OUT>: out = AO@Wo^T+bo -> d_out fp32
// GEMM wave tile is 32(M) x 64(N): 8 WMMAs per k-step off 6 fragment loads
// (24 B/lane per WMMA vs 40 at 16x64) -- better compute density for the
// compute-bound projections.
// ---------------------------------------------------------------------------

typedef __attribute__((ext_vector_type(16))) __bf16 v16bf;
typedef __attribute__((ext_vector_type(8)))  __bf16 v8bf;
typedef __attribute__((ext_vector_type(8)))  float  v8f;

#define D_MODEL 1024
#define S_LEN   2048
#define NHEAD   16
#define HDIM    64
#define NTOK    4096          /* B * S */

__device__ inline v16bf cat8(v8bf a, v8bf b) {
    return __builtin_shufflevector(a, b, 0,1,2,3,4,5,6,7,8,9,10,11,12,13,14,15);
}
__device__ inline v8bf cvt8(v8f x) {
    v8bf r;
#pragma unroll
    for (int i = 0; i < 8; ++i) r[i] = (__bf16)x[i];
    return r;
}
__device__ inline v8f wmma_bf16(v16bf a, v16bf b, v8f c) {
    return __builtin_amdgcn_wmma_f32_16x16x32_bf16(
        /*neg_a=*/false, a, /*neg_b=*/false, b,
        /*c_mod=*/(short)0, c, /*reuse_a=*/false, /*reuse_b=*/false);
}

// -------------------------- fp32 -> bf16 convert ---------------------------
__global__ void cvt_f32_bf16(const float* __restrict__ src,
                             __bf16* __restrict__ dst, int n) {
    int i = blockIdx.x * blockDim.x + threadIdx.x;
    if (i < n) dst[i] = (__bf16)src[i];
}

// ------------------------------ WMMA GEMM ----------------------------------
// C[M=4096][N=1024] = A[4096][1024] @ W^T + bias ; W is [1024 out][1024 in].
// One wave computes a 32(M) x 64(N) tile: 2 A frags x 4 B frags -> 8 WMMAs.
// MODE: 0 = bf16 out, per-head row-major Qh/Kh ; 1 = bf16 out, transposed Vt ;
//       2 = fp32 out row-major (final projection).  ABF: A is bf16 (else f32).
template<int MODE, bool ABF>
__global__ __launch_bounds__(128)
void gemm_wmma(const void* __restrict__ Aptr, const __bf16* __restrict__ W,
               const float* __restrict__ bias, void* __restrict__ Dst) {
    const int lane = threadIdx.x & 31;
    const int warp = threadIdx.x >> 5;
    const int wid  = blockIdx.x * 4 + warp;       // 2048 waves total
    const int m0   = (wid >> 4) << 5;             // 128 M tiles of 32 rows
    const int n0   = (wid & 15) << 6;             // 16 N super-tiles of 64
    const int lm   = lane & 15;
    const int hsel = (lane >> 4) & 1;             // lane half
    const int koffA = hsel * 8;                   // A-frag lane K offset

    v8f acc[2][4] = {};
    const __bf16* Ab = (const __bf16*)Aptr;
    const float*  Af = (const float*)Aptr;

    for (int k = 0; k < D_MODEL; k += 32) {
        v16bf a[2];
#pragma unroll
        for (int mi = 0; mi < 2; ++mi) {
            const size_t row = (size_t)(m0 + mi * 16 + lm);
            if constexpr (ABF) {
                const __bf16* ap = Ab + row * D_MODEL + k + koffA;
                a[mi] = cat8(*(const v8bf*)ap, *(const v8bf*)(ap + 16));
            } else {
                const float* ap = Af + row * D_MODEL + k + koffA;
                a[mi] = cat8(cvt8(*(const v8f*)ap), cvt8(*(const v8f*)(ap + 16)));
            }
        }
        const int kinB = k + hsel * 8;            // B-frag lane K offset
#pragma unroll
        for (int j = 0; j < 4; ++j) {
            const int nout = n0 + j * 16 + lm;    // B lane = output feature
            const __bf16* wp = W + (size_t)nout * D_MODEL + kinB;
            v16bf b = cat8(*(const v8bf*)wp, *(const v8bf*)(wp + 16));
            acc[0][j] = wmma_bf16(a[0], b, acc[0][j]);
            acc[1][j] = wmma_bf16(a[1], b, acc[1][j]);
        }
    }

    // Epilogue: C layout = VGPR r -> row M=r+8*hsel, lane lm -> col within tile
#pragma unroll
    for (int mi = 0; mi < 2; ++mi) {
        const int mb = m0 + mi * 16;
#pragma unroll
        for (int j = 0; j < 4; ++j) {
            const int nf = n0 + j * 16 + lm;
            const float bv = bias[nf];
            if constexpr (MODE == 2) {            // fp32 row-major out
                float* out = (float*)Dst;
#pragma unroll
                for (int r = 0; r < 8; ++r) {
                    int t = mb + hsel * 8 + r;
                    out[(size_t)t * D_MODEL + nf] = acc[mi][j][r] + bv;
                }
            } else if constexpr (MODE == 0) {     // Qh/Kh[bh][s][d] bf16
                __bf16* out = (__bf16*)Dst;
                const int h = nf >> 6, d = nf & 63;
#pragma unroll
                for (int r = 0; r < 8; ++r) {
                    int t = mb + hsel * 8 + r;
                    int bb = t >> 11, s = t & (S_LEN - 1);
                    out[(((size_t)(bb * NHEAD + h)) * S_LEN + s) * HDIM + d] =
                        (__bf16)(acc[mi][j][r] + bv);
                }
            } else {                              // Vt[bh][d][s] bf16, packed
                __bf16* out = (__bf16*)Dst;
                const int h = nf >> 6, d = nf & 63;
                const int t0 = mb + hsel * 8;
                const int bb = t0 >> 11, s0 = t0 & (S_LEN - 1);
                v8bf pk;
#pragma unroll
                for (int r = 0; r < 8; ++r) pk[r] = (__bf16)(acc[mi][j][r] + bv);
                *(v8bf*)(out + ((size_t)(bb * NHEAD + h) * HDIM + d) * S_LEN + s0) = pk;
            }
        }
    }
}

// --------------------------- causal attention ------------------------------
// One wave per (bh, 16-query tile). Online softmax in f32; scores and PV via
// WMMA. P is bounced through per-wave LDS to convert C-layout -> A-layout
// (same-wave DS ops are in-order, so no barrier needed).
__global__ __launch_bounds__(128)
void attn_wmma(const __bf16* __restrict__ Qh, const __bf16* __restrict__ Kh,
               const __bf16* __restrict__ Vt, __bf16* __restrict__ AO) {
    __shared__ __align__(32) __bf16 pbuf[4][16][32];

    const int lane = threadIdx.x & 31;
    const int warp = threadIdx.x >> 5;
    const int wid  = blockIdx.x * 4 + warp;       // 4096 waves
    const int qt   = wid & 127;                   // S/16 query tiles
    const int bh   = wid >> 7;                    // 0..31
    const int q0   = qt << 4;
    const int lm   = lane & 15;
    const int hsel = (lane >> 4) & 1;
    const float SC = 0.125f;                      // HEAD_DIM^-0.5

    // Q fragments for the whole tile (d = 0..31 and 32..63), reused all loop.
    const __bf16* qbase = Qh + ((size_t)bh * S_LEN + (q0 + lm)) * HDIM;
    v16bf qa[2];
#pragma unroll
    for (int s = 0; s < 2; ++s) {
        const int ko = s * 32 + hsel * 8;
        qa[s] = cat8(*(const v8bf*)(qbase + ko), *(const v8bf*)(qbase + ko + 16));
    }

    float mrow[8], lrow[8];
    v8f oacc[4] = {};
#pragma unroll
    for (int r = 0; r < 8; ++r) { mrow[r] = -3.0e38f; lrow[r] = 0.0f; }

    const int kbmax = q0 >> 5;                    // causal: key blocks of 32
    for (int kb = 0; kb <= kbmax; ++kb) {
        const int key0 = kb << 5;

        // S = Q K^T for two 16-key tiles (K-dim = head dim, 2 wmma each)
        v8f sc[2];
#pragma unroll
        for (int t = 0; t < 2; ++t) {
            const int key = key0 + t * 16 + lm;   // B lane = key position
            const __bf16* kp =
                Kh + ((size_t)bh * S_LEN + key) * HDIM + hsel * 8;
            v8f z = {};
            z = wmma_bf16(qa[0], cat8(*(const v8bf*)kp, *(const v8bf*)(kp + 16)), z);
            z = wmma_bf16(qa[1], cat8(*(const v8bf*)(kp + 32), *(const v8bf*)(kp + 48)), z);
            sc[t] = z;
        }

        // scale + causal mask (only diagonal block needs masking)
        const bool diag = (kb == kbmax);
#pragma unroll
        for (int t = 0; t < 2; ++t) {
            const int key = key0 + t * 16 + lm;
#pragma unroll
            for (int r = 0; r < 8; ++r) {
                float v = sc[t][r] * SC;
                if (diag && key > (q0 + hsel * 8 + r)) v = -3.0e38f;
                sc[t][r] = v;
            }
        }

        // Online softmax: rows live across 16-lane halves of one VGPR.
        float p0[8], p1[8], al[8];
#pragma unroll
        for (int r = 0; r < 8; ++r) {
            float mx = fmaxf(sc[0][r], sc[1][r]);
            for (int off = 1; off < 16; off <<= 1)
                mx = fmaxf(mx, __shfl_xor(mx, off));
            const float mnew = fmaxf(mrow[r], mx);
            const float a = __expf(mrow[r] - mnew);
            p0[r] = __expf(sc[0][r] - mnew);
            p1[r] = __expf(sc[1][r] - mnew);
            float srow = p0[r] + p1[r];
            for (int off = 1; off < 16; off <<= 1)
                srow += __shfl_xor(srow, off);
            lrow[r] = lrow[r] * a + srow;
            mrow[r] = mnew;
            al[r] = a;
        }
#pragma unroll
        for (int j = 0; j < 4; ++j)
#pragma unroll
            for (int r = 0; r < 8; ++r) oacc[j][r] *= al[r];

        // C-layout P -> LDS (row-major 16x32) -> A-layout fragment.
#pragma unroll
        for (int r = 0; r < 8; ++r) {
            pbuf[warp][hsel * 8 + r][lm]      = (__bf16)p0[r];
            pbuf[warp][hsel * 8 + r][16 + lm] = (__bf16)p1[r];
        }
        const __bf16* pb = &pbuf[warp][lm][hsel * 8];
        v16bf pa = cat8(*(const v8bf*)pb, *(const v8bf*)(pb + 16));

        // O += P V  (B from transposed Vt: lane = head-dim column, contiguous)
#pragma unroll
        for (int j = 0; j < 4; ++j) {
            const __bf16* vp = Vt + ((size_t)bh * HDIM + j * 16 + lm) * S_LEN +
                               key0 + hsel * 8;
            oacc[j] = wmma_bf16(pa, cat8(*(const v8bf*)vp, *(const v8bf*)(vp + 16)), oacc[j]);
        }
    }

    // Normalize and scatter back to [token][h*64+d] bf16 for the O-projection.
    const int bb = bh >> 4, h = bh & 15;
#pragma unroll
    for (int j = 0; j < 4; ++j) {
        const int feat = h * 64 + j * 16 + lm;
#pragma unroll
        for (int r = 0; r < 8; ++r) {
            const int s = q0 + hsel * 8 + r;
            AO[((size_t)(bb * S_LEN + s)) * D_MODEL + feat] =
                (__bf16)(oacc[j][r] / lrow[r]);
        }
    }
}

// ------------------------------- launcher ----------------------------------
extern "C" void kernel_launch(void* const* d_in, const int* in_sizes, int n_in,
                              void* d_out, int out_size, void* d_ws, size_t ws_size,
                              hipStream_t stream) {
    (void)in_sizes; (void)n_in; (void)out_size; (void)ws_size;
    const float* q  = (const float*)d_in[0];
    const float* k  = (const float*)d_in[1];
    const float* v  = (const float*)d_in[2];
    const float* Wq = (const float*)d_in[3];  const float* bq = (const float*)d_in[4];
    const float* Wk = (const float*)d_in[5];  const float* bk = (const float*)d_in[6];
    const float* Wv = (const float*)d_in[7];  const float* bv = (const float*)d_in[8];
    const float* Wo = (const float*)d_in[9];  const float* bo = (const float*)d_in[10];

    const size_t WEL = (size_t)D_MODEL * D_MODEL;      // 1M weight elems
    const size_t TEL = (size_t)NTOK * D_MODEL;         // 4M activation elems
    __bf16* wqb = (__bf16*)d_ws;
    __bf16* wkb = wqb + WEL;
    __bf16* wvb = wkb + WEL;
    __bf16* wob = wvb + WEL;
    __bf16* Qh  = wob + WEL;
    __bf16* Kh  = Qh + TEL;
    __bf16* Vt  = Kh + TEL;
    __bf16* AO  = Vt + TEL;

    const int cb = (int)(WEL / 256);
    cvt_f32_bf16<<<cb, 256, 0, stream>>>(Wq, wqb, (int)WEL);
    cvt_f32_bf16<<<cb, 256, 0, stream>>>(Wk, wkb, (int)WEL);
    cvt_f32_bf16<<<cb, 256, 0, stream>>>(Wv, wvb, (int)WEL);
    cvt_f32_bf16<<<cb, 256, 0, stream>>>(Wo, wob, (int)WEL);

    // 2048 waves per GEMM (32x64 tiles), 4 waves per block -> 512 blocks.
    gemm_wmma<0, false><<<512, 128, 0, stream>>>(q, wqb, bq, Qh);
    gemm_wmma<0, false><<<512, 128, 0, stream>>>(k, wkb, bk, Kh);
    gemm_wmma<1, false><<<512, 128, 0, stream>>>(v, wvb, bv, Vt);

    attn_wmma<<<1024, 128, 0, stream>>>(Qh, Kh, Vt, AO);

    gemm_wmma<2, true><<<512, 128, 0, stream>>>(AO, wob, bo, d_out);
}
